// VectorQuantizer_80264348828255
// MI455X (gfx1250) — matrix operand order repaired and checked by reading, not compiled, not run on hardware
//
#include <hip/hip_runtime.h>
#include <hip/hip_fp16.h>

typedef __attribute__((ext_vector_type(16))) _Float16 v16h;
typedef __attribute__((ext_vector_type(8)))  float    v8f;

#define N_ROWS        32768
#define DDIM          256
#define KCB           8192
#define ROWS_PER_BLK  128      // 4 waves x 2 tiles x 16 rows
#define STAGE_ROWS    64       // LDS staging pass size
#define XS_STRIDE     264      // halfs, padded to dodge bank conflicts
#define NCHUNKS       (KCB / 16)
#define CHUNK_BYTES   8192     // 8 dsteps * 32 lanes * 16 halfs * 2B
#define BUF_HALFS     4096

// ---------------------------------------------------------------------------
// Repack codebook E [D=256][K=8192] f32 -> f16 in WMMA B-fragment order.
// Fragment for (chunk c, dstep t): lane l holds column n = 16c + (l&15),
// kgroup = l>>4; VGPR v packs K = 32t + 16*kgroup + 2v (lo half) and +1 (hi).
// Stored so each lane reads its 16 halfs (32B) contiguously.
// ---------------------------------------------------------------------------
__global__ void vq_pack_e(const float* __restrict__ E, _Float16* __restrict__ epack) {
    int tid  = blockIdx.x * blockDim.x + threadIdx.x;   // 0 .. 512*8*32-1
    int lane = tid & 31;
    int t    = (tid >> 5) & 7;
    int c    = tid >> 8;
    int n    = (c << 4) + (lane & 15);
    int kg   = lane >> 4;
    __align__(16) _Float16 vals[16];
#pragma unroll
    for (int v = 0; v < 8; ++v) {
        int kd = 32 * t + 16 * kg + 2 * v;
        vals[2 * v]     = (_Float16)E[(size_t)kd * KCB + n];
        vals[2 * v + 1] = (_Float16)E[(size_t)(kd + 1) * KCB + n];
    }
    uint4* dst = (uint4*)(epack + (size_t)tid * 16);
    dst[0] = *(const uint4*)&vals[0];
    dst[1] = *(const uint4*)&vals[8];
}

// ||e_k||^2 in f32 (exact inputs; argmin bias term)
__global__ void vq_enorm(const float* __restrict__ E, float* __restrict__ enorm) {
    int k = blockIdx.x * blockDim.x + threadIdx.x;
    float s = 0.f;
    for (int d = 0; d < DDIM; ++d) {
        float v = E[(size_t)d * KCB + k];
        s = fmaf(v, v, s);
    }
    enorm[k] = s;
}

// Pull a 16x32-per-dstep A fragment set (contiguous k = 16*hi + s order, same as packed B) from staged LDS.
__device__ __forceinline__ void load_afrags(const _Float16* xs, int rowInStage,
                                            int m, int hi, v16h* afrag) {
#pragma unroll
    for (int t = 0; t < 8; ++t) {
#pragma unroll
        for (int v = 0; v < 8; ++v) {
            int kb = 32 * t + 16 * hi + 2 * v;
            ((unsigned*)&afrag[t])[v] =
                *(const unsigned*)&xs[(rowInStage + m) * XS_STRIDE + kb];
        }
    }
}

// Async-copy one 8KB packed-B chunk into an LDS buffer (128 threads, 16B each
// x4). GV mode: %0 = LDS byte address VGPR, %1 = 64-bit global address pair.
__device__ __forceinline__ void async_copy_chunk(const char* epbase, int c,
                                                 unsigned ldsDst, int tid) {
    unsigned lofs = (unsigned)tid * 16u;
    size_t   gofs = (size_t)c * CHUNK_BYTES + lofs;
#pragma unroll
    for (int j = 0; j < 4; ++j) {
        unsigned long long ga = (unsigned long long)(size_t)epbase + gofs + (unsigned)(j * 2048);
        unsigned           la = ldsDst + lofs + (unsigned)(j * 2048);
        asm volatile("global_load_async_to_lds_b128 %0, %1, off"
                     :: "v"(la), "v"(ga) : "memory");
    }
}

// ---------------------------------------------------------------------------
// Fused score-GEMM + argmin. 4 waves/block; each wave owns two 16-row tiles.
// Packed B chunks are double-buffered in LDS via async global->LDS copies
// (ASYNCcnt), so the WMMA stream never waits on an L2 round trip: chunk c+1
// streams in while chunk c is consumed from LDS.
// ---------------------------------------------------------------------------
__global__ __launch_bounds__(128) void vq_argmin(
    const float* __restrict__ X, const _Float16* __restrict__ epack,
    const float* __restrict__ enorm, int* __restrict__ idx) {
    __shared__ _Float16 xs[STAGE_ROWS * XS_STRIDE];        // 33,792 B
    __shared__ __align__(16) _Float16 bbuf[2 * BUF_HALFS]; // 16,384 B

    const int tid      = threadIdx.x;
    const int lane     = tid & 31;
    const int wave     = tid >> 5;
    const int rowBlock = blockIdx.x * ROWS_PER_BLK;
    const int m        = lane & 15;   // A row / C column within tile
    const int hi       = lane >> 4;

    v16h afrag0[8], afrag1[8];

    // Stage pass 0: rows [rowBlock, rowBlock+64) -> f16 LDS, grab tile-0 frags
    for (int i = tid; i < STAGE_ROWS * DDIM; i += 128) {
        int r = i >> 8, col = i & 255;
        xs[r * XS_STRIDE + col] = (_Float16)X[(size_t)(rowBlock + r) * DDIM + col];
    }
    __syncthreads();
    load_afrags(xs, wave * 16, m, hi, afrag0);
    __syncthreads();

    // Stage pass 1: rows [rowBlock+64, rowBlock+128), grab tile-1 frags
    for (int i = tid; i < STAGE_ROWS * DDIM; i += 128) {
        int r = i >> 8, col = i & 255;
        xs[r * XS_STRIDE + col] =
            (_Float16)X[(size_t)(rowBlock + STAGE_ROWS + r) * DDIM + col];
    }
    __syncthreads();
    load_afrags(xs, wave * 16, m, hi, afrag1);

    float minv0[8], minv1[8];
    int   mini0[8], mini1[8];
#pragma unroll
    for (int e = 0; e < 8; ++e) {
        minv0[e] = 3.4e38f; mini0[e] = 0;
        minv1[e] = 3.4e38f; mini1[e] = 0;
    }

    const char*    epbase = (const char*)epack;
    const unsigned ldsB0  = (unsigned)(size_t)&bbuf[0];
    const unsigned ldsB1  = (unsigned)(size_t)&bbuf[BUF_HALFS];

    // Prologue: start streaming chunk 0 into buffer 0
    async_copy_chunk(epbase, 0, ldsB0, tid);

    for (int c = 0; c < NCHUNKS; ++c) {
        // Chunk c's copy (issued one full chunk of WMMAs ago) must be done;
        // barrier makes all 4 waves' pieces visible and fences last chunk's
        // LDS reads before that buffer is overwritten below.
        asm volatile("s_wait_asynccnt 0x0" ::: "memory");
        __syncthreads();
        if (c + 1 < NCHUNKS)
            async_copy_chunk(epbase, c + 1, ((c + 1) & 1) ? ldsB1 : ldsB0, tid);

        const _Float16* bb = (c & 1) ? &bbuf[BUF_HALFS] : &bbuf[0];
        float en = enorm[c * 16 + m];
        v8f acc0 = {}, acc1 = {};
#pragma unroll
        for (int t = 0; t < 8; ++t) {
            v16h b;
            const uint4* p = (const uint4*)&bb[(t * 32 + lane) * 16];
            ((uint4*)&b)[0] = p[0];
            ((uint4*)&b)[1] = p[1];
            acc0 = __builtin_amdgcn_wmma_f32_16x16x32_f16(
                false, afrag0[t], false, b, (short)0, acc0, false, false);
            acc1 = __builtin_amdgcn_wmma_f32_16x16x32_f16(
                false, afrag1[t], false, b, (short)0, acc1, false, false);
        }
        int col = c * 16 + m;
#pragma unroll
        for (int e = 0; e < 8; ++e) {
            float d0 = fmaf(-2.f, acc0[e], en);   // ||e||^2 - 2 x.e
            float d1 = fmaf(-2.f, acc1[e], en);
            if (d0 < minv0[e]) { minv0[e] = d0; mini0[e] = col; }
            if (d1 < minv1[e]) { minv1[e] = d1; mini1[e] = col; }
        }
    }

    // Reduce across the 16 lanes holding one row (C layout: element e is
    // row e + 8*hi; lanes 0-15 and 16-31 are independent row groups).
#pragma unroll
    for (int e = 0; e < 8; ++e) {
        float v0 = minv0[e]; int i0 = mini0[e];
        float v1 = minv1[e]; int i1 = mini1[e];
#pragma unroll
        for (int s = 1; s < 16; s <<= 1) {
            float ov0 = __shfl_xor(v0, s, 16);
            int   oi0 = __shfl_xor(i0, s, 16);
            float ov1 = __shfl_xor(v1, s, 16);
            int   oi1 = __shfl_xor(i1, s, 16);
            if (ov0 < v0 || (ov0 == v0 && oi0 < i0)) { v0 = ov0; i0 = oi0; }
            if (ov1 < v1 || (ov1 == v1 && oi1 < i1)) { v1 = ov1; i1 = oi1; }
        }
        if (m == 0) {
            int r = wave * 16 + hi * 8 + e;
            idx[rowBlock + r]              = i0;
            idx[rowBlock + STAGE_ROWS + r] = i1;
        }
    }
}

// Gather f32 codewords (exact output) + fixed-order partial loss sums
__global__ void vq_gather(const float* __restrict__ X, const float* __restrict__ E,
                          const int* __restrict__ idx, float* __restrict__ out,
                          float* __restrict__ partials) {
    __shared__ float red[256];
    int tid = threadIdx.x;
    size_t base = (size_t)blockIdx.x * 1024;
    float s = 0.f;
#pragma unroll
    for (int j = 0; j < 4; ++j) {
        size_t i = base + tid + (size_t)j * 256;
        int n = (int)(i >> 8);
        int d = (int)(i & 255);
        float q = E[(size_t)d * KCB + idx[n]];
        float x = X[i];
        out[i] = q;                       // x + sg(q - x) == q numerically
        float df = q - x;
        s = fmaf(df, df, s);
    }
    red[tid] = s;
    __syncthreads();
    for (int st = 128; st > 0; st >>= 1) {
        if (tid < st) red[tid] += red[tid + st];
        __syncthreads();
    }
    if (tid == 0) partials[blockIdx.x] = red[0];
}

__global__ void vq_finalize(const float* __restrict__ partials, float* __restrict__ loss_out) {
    __shared__ float red[256];
    int tid = threadIdx.x;
    float s = 0.f;
    for (int j = tid; j < 8192; j += 256) s += partials[j];   // fixed order
    red[tid] = s;
    __syncthreads();
    for (int st = 128; st > 0; st >>= 1) {
        if (tid < st) red[tid] += red[tid + st];
        __syncthreads();
    }
    if (tid == 0) loss_out[0] = 1.25f * red[0] / 8388608.f;   // (1+BETA)*mean
}

extern "C" void kernel_launch(void* const* d_in, const int* in_sizes, int n_in,
                              void* d_out, int out_size, void* d_ws, size_t ws_size,
                              hipStream_t stream) {
    (void)in_sizes; (void)n_in; (void)out_size; (void)ws_size;
    const float* X = (const float*)d_in[0];
    const float* E = (const float*)d_in[1];
    float* out = (float*)d_out;

    char* ws = (char*)d_ws;
    _Float16* epack    = (_Float16*)ws;                                   // 4 MB
    float*    enorm    = (float*)(ws + (size_t)4 * 1024 * 1024);          // 32 KB
    int*      idx      = (int*)(ws + (size_t)4 * 1024 * 1024 + 32 * 1024);// 128 KB
    float*    partials = (float*)(ws + (size_t)4 * 1024 * 1024 + 160 * 1024); // 32 KB

    vq_pack_e  <<<512, 256, 0, stream>>>(E, epack);
    vq_enorm   <<<KCB / 256, 256, 0, stream>>>(E, enorm);
    vq_argmin  <<<N_ROWS / ROWS_PER_BLK, 128, 0, stream>>>(X, epack, enorm, idx);
    vq_gather  <<<8192, 256, 0, stream>>>(X, E, idx, out, partials);
    vq_finalize<<<1, 256, 0, stream>>>(partials, out + (size_t)N_ROWS * DDIM);
}
